// AliSwin3D_45767171506342
// MI455X (gfx1250) — matrix operand order repaired and checked by reading, not compiled, not run on hardware
//
#include <hip/hip_runtime.h>
#include <hip/hip_bf16.h>
#include <math.h>

// ---------------------------------------------------------------------------
// CDNA5 (gfx1250) Swin-3D forward. All GEMMs + attention use
// v_wmma_f32_16x16x32_f16 (f16 A/B, f32 accumulate), wave32.
// Fragment layouts per CDNA5 ISA 7.12.2:
//   A 16x32 f16 : lane row = lane&15, half = lane>>4;
//                 elems 0..7  <-> K = half*8 + e        (contiguous)
//                 elems 8..15 <-> K = 16 + half*8 + e-8 (contiguous)
//   B 32x16 f16 : lane col = lane&15; elems e <-> same K as A (half*8 + e, 16 + half*8 + e-8)
//   C/D 16x16 f32: vgpr v -> row v + 8*half, col = lane&15
// LDS tiles are laid out so each fragment is exactly two ds_load_b128.
// Staging issues all global b128 loads before converting (loads in flight).
// ---------------------------------------------------------------------------

typedef __attribute__((ext_vector_type(16))) _Float16 v16h;
typedef __attribute__((ext_vector_type(8)))  _Float16 v8h;
typedef __attribute__((ext_vector_type(8)))  float    v8f;
typedef __attribute__((ext_vector_type(4)))  float    v4f;

union AFrag { v16h v; v8h g[2]; _Float16 h[16]; };
union CFrag { v8f  v; float f[8]; };

__device__ __forceinline__ float gelu_exact(float x) {
    return 0.5f * x * (1.0f + erff(x * 0.70710678118654752f));
}

// ---------------------------------------------------------------------------
// GEMM: C[M,N] = act(A[M,K] @ B[K,N] + bias) + residual
// REQUIRES: M % 64 == 0, N % 32 == 0, K % 32 == 0 (true for every call site).
// Block = 128 threads (4 waves). Block tile 64x32; wave w -> rows [16w,16w+16).
// ---------------------------------------------------------------------------
__global__ __launch_bounds__(128) void gemm_wmma(
    const float* __restrict__ A, const float* __restrict__ B,
    const float* __restrict__ bias, const float* __restrict__ residual,
    float* __restrict__ C, int M, int N, int K, int act)
{
    __shared__ _Float16 As [64][40];   // row-major A tile, 80B rows (16B aligned)
    __shared__ _Float16 BsT[32][40];   // B tile transposed: BsT[col][k]

    const int tid  = threadIdx.x;
    const int lane = tid & 31;
    const int wid  = tid >> 5;
    const int m0   = blockIdx.y * 64;
    const int n0   = blockIdx.x * 32;
    const int wm   = wid * 16;
    const int row  = lane & 15;
    const int half = lane >> 4;

    // per-thread staging coordinates (constant across k tiles)
    const int ar_  = tid >> 1;              // A row 0..63
    const int ac4  = (tid & 1) << 4;        // A col group {0,16}: 4 float4 each
    const int br_  = tid >> 2;              // B k-row 0..31
    const int bc4  = (tid & 3) << 3;        // B col group {0,8,16,24}: 2 float4

    CFrag c0, c1;
    #pragma unroll
    for (int i = 0; i < 8; ++i) { c0.f[i] = 0.0f; c1.f[i] = 0.0f; }

    const int kT = K >> 5;
    for (int kt = 0; kt < kT; ++kt) {
        const int k0 = kt << 5;

        // ---- issue ALL staging loads first (keeps loads in flight) ----
        v4f av[4], bv[2];
        {
            const float* ap = A + (size_t)(m0 + ar_) * K + k0 + ac4;
            #pragma unroll
            for (int it = 0; it < 4; ++it) av[it] = *(const v4f*)(ap + 4 * it);
            const float* bp = B + (size_t)(k0 + br_) * N + n0 + bc4;
            #pragma unroll
            for (int it = 0; it < 2; ++it) bv[it] = *(const v4f*)(bp + 4 * it);
        }
        // ---- convert + store to LDS ----
        #pragma unroll
        for (int it = 0; it < 4; ++it) {
            _Float16* d = &As[ar_][ac4 + 4 * it];
            d[0] = (_Float16)av[it][0]; d[1] = (_Float16)av[it][1];
            d[2] = (_Float16)av[it][2]; d[3] = (_Float16)av[it][3];
        }
        #pragma unroll
        for (int it = 0; it < 2; ++it) {
            const int cb = bc4 + 4 * it;
            BsT[cb + 0][br_] = (_Float16)bv[it][0];
            BsT[cb + 1][br_] = (_Float16)bv[it][1];
            BsT[cb + 2][br_] = (_Float16)bv[it][2];
            BsT[cb + 3][br_] = (_Float16)bv[it][3];
        }
        if (kt + 1 < kT) // hint next A tile into cache (global_prefetch_b8)
            __builtin_prefetch(A + (size_t)(m0 + lane) * K + k0 + 32, 0, 1);
        __syncthreads();

        AFrag a, b0, b1;
        const _Float16* ar  = &As[wm + row][0];
        a.g[0]  = *(const v8h*)(ar + half * 8);
        a.g[1]  = *(const v8h*)(ar + 16 + half * 8);
        const _Float16* br0 = &BsT[row][0];
        b0.g[0] = *(const v8h*)(br0 + half * 8);
        b0.g[1] = *(const v8h*)(br0 + 16 + half * 8);
        const _Float16* br1 = &BsT[16 + row][0];
        b1.g[0] = *(const v8h*)(br1 + half * 8);
        b1.g[1] = *(const v8h*)(br1 + 16 + half * 8);

        c0.v = __builtin_amdgcn_wmma_f32_16x16x32_f16(
            false, a.v, false, b0.v, (short)0, c0.v, false, false);
        c1.v = __builtin_amdgcn_wmma_f32_16x16x32_f16(
            false, a.v, false, b1.v, (short)0, c1.v, false, false);
        __syncthreads();
    }

    #pragma unroll
    for (int v = 0; v < 8; ++v) {
        const int gr = m0 + wm + v + 8 * half;
        const size_t base = (size_t)gr * N + n0;
        {
            float val = c0.f[v];
            if (bias) val += bias[n0 + row];
            if (act == 1) val = gelu_exact(val);
            if (residual) val += residual[base + row];
            C[base + row] = val;
        }
        {
            float val = c1.f[v];
            if (bias) val += bias[n0 + 16 + row];
            if (act == 1) val = gelu_exact(val);
            if (residual) val += residual[base + 16 + row];
            C[base + 16 + row] = val;
        }
    }
}

// ---------------------------------------------------------------------------
// Windowed 3D attention, window (4,8,8) -> N=256 tokens, head dim 32.
// One wave per (window, head, 16-row tile). QK^T: 16 WMMAs (K=32 each),
// softmax + rel-pos-bias + shift mask in LDS, PV: 16 WMMAs.
// Roll / window partition / reverse folded into index math.
// qkv re-reads across the 16 row tiles stay resident in the 192MB L2.
// ---------------------------------------------------------------------------
__device__ __forceinline__ int tok_index(int n, int wdi, int whi, int wwi,
                                         int Dd, int Hh, int Ww,
                                         int sD, int sH, int sW)
{
    const int ld = n >> 6, lh = (n >> 3) & 7, lw = n & 7;
    int d = wdi * 4 + ld + sD; if (d >= Dd) d -= Dd;
    int h = whi * 8 + lh + sH; if (h >= Hh) h -= Hh;
    int w = wwi * 8 + lw + sW; if (w >= Ww) w -= Ww;
    return (d * Hh + h) * Ww + w;
}

__device__ __forceinline__ int region_of(int n, int wdi, int whi, int wwi,
                                         int Dd, int Hh, int Ww,
                                         int sD, int sH, int sW)
{
    const int gd = wdi * 4 + (n >> 6);
    const int gh = whi * 8 + ((n >> 3) & 7);
    const int gw = wwi * 8 + (n & 7);
    const int rd = (sD == 0) ? 0 : (gd < Dd - 4 ? 0 : (gd < Dd - sD ? 1 : 2));
    const int rh = (sH == 0) ? 0 : (gh < Hh - 8 ? 0 : (gh < Hh - sH ? 1 : 2));
    const int rw = (sW == 0) ? 0 : (gw < Ww - 8 ? 0 : (gw < Ww - sW ? 1 : 2));
    return (rd * 3 + rh) * 3 + rw;
}

__global__ __launch_bounds__(32) void attn_win(
    const float* __restrict__ qkv,   // [T, 3C], 3C dim = [which][head][32]
    const float* __restrict__ rpb,   // [1575, NH]
    float* __restrict__ out,         // [T, C]
    int Dd, int Hh, int Ww, int C, int NH, int sD, int sH, int sW)
{
    __shared__ _Float16 Ks [256][32];   // K row-major: rows are K-contiguous
    __shared__ _Float16 VsT[32][256];   // V transposed: VsT[dim][token]
    __shared__ _Float16 Qs [16][32];
    __shared__ float    Sc [16][256];
    __shared__ _Float16 Ps [16][256];
    __shared__ int      Rg [256];

    const int bid  = blockIdx.x;
    const int rt   = bid & 15;            // row tile (N=256 -> 16 tiles of 16)
    const int head = (bid >> 4) % NH;
    const int wid  = (bid >> 4) / NH;
    const int nWh = Hh >> 3, nWw = Ww >> 3;
    const int wdi = wid / (nWh * nWw);
    const int rem = wid % (nWh * nWw);
    const int whi = rem / nWw, wwi = rem % nWw;

    const int lane = threadIdx.x;
    const size_t c3 = (size_t)3 * C;
    const bool masked = (sD | sH | sW) != 0;

    // load K,V (256x32 each): issue all 16 b128 loads per token, then convert
    for (int n = lane; n < 256; n += 32) {
        const int t = tok_index(n, wdi, whi, wwi, Dd, Hh, Ww, sD, sH, sW);
        Rg[n] = region_of(n, wdi, whi, wwi, Dd, Hh, Ww, sD, sH, sW);
        const float* kp = qkv + (size_t)t * c3 + C + head * 32;
        const float* vp = kp + C;
        v4f kv[8], vv[8];
        #pragma unroll
        for (int g = 0; g < 8; ++g) kv[g] = *(const v4f*)(kp + g * 4);
        #pragma unroll
        for (int g = 0; g < 8; ++g) vv[g] = *(const v4f*)(vp + g * 4);
        #pragma unroll
        for (int g = 0; g < 8; ++g) {
            _Float16* kd = &Ks[n][g * 4];
            kd[0] = (_Float16)kv[g][0]; kd[1] = (_Float16)kv[g][1];
            kd[2] = (_Float16)kv[g][2]; kd[3] = (_Float16)kv[g][3];
            VsT[g * 4 + 0][n] = (_Float16)vv[g][0];
            VsT[g * 4 + 1][n] = (_Float16)vv[g][1];
            VsT[g * 4 + 2][n] = (_Float16)vv[g][2];
            VsT[g * 4 + 3][n] = (_Float16)vv[g][3];
        }
    }
    if (lane < 16) { // this tile's 16 query rows, pre-scaled by hd^-0.5
        const int nq = rt * 16 + lane;
        const int t = tok_index(nq, wdi, whi, wwi, Dd, Hh, Ww, sD, sH, sW);
        const float* qp = qkv + (size_t)t * c3 + head * 32;
        v4f qv[8];
        #pragma unroll
        for (int g = 0; g < 8; ++g) qv[g] = *(const v4f*)(qp + g * 4);
        #pragma unroll
        for (int g = 0; g < 8; ++g) {
            _Float16* qd = &Qs[lane][g * 4];
            qd[0] = (_Float16)(qv[g][0] * 0.17677669529663689f);
            qd[1] = (_Float16)(qv[g][1] * 0.17677669529663689f);
            qd[2] = (_Float16)(qv[g][2] * 0.17677669529663689f);
            qd[3] = (_Float16)(qv[g][3] * 0.17677669529663689f);
        }
    }
    __syncthreads();

    const int row = lane & 15, half = lane >> 4;

    // Q a-fragment (invariant across column tiles): two ds_load_b128
    AFrag qa;
    {
        const _Float16* qr = &Qs[row][0];
        qa.g[0] = *(const v8h*)(qr + half * 8);
        qa.g[1] = *(const v8h*)(qr + 16 + half * 8);
    }

    // scores = Q @ K^T : one 16x16x32 WMMA per 16-col tile, +rpb +mask
    for (int ct = 0; ct < 16; ++ct) {
        AFrag b;
        const _Float16* kr = &Ks[ct * 16 + row][0];
        b.g[0] = *(const v8h*)(kr + half * 8);
        b.g[1] = *(const v8h*)(kr + 16 + half * 8);

        CFrag s;
        #pragma unroll
        for (int i = 0; i < 8; ++i) s.f[i] = 0.0f;
        s.v = __builtin_amdgcn_wmma_f32_16x16x32_f16(
            false, qa.v, false, b.v, (short)0, s.v, false, false);

        #pragma unroll
        for (int v = 0; v < 8; ++v) {
            const int m  = v + 8 * half;
            const int nq = rt * 16 + m;
            const int nk = ct * 16 + row;
            const int dd = (nq >> 6) - (nk >> 6);
            const int dh = ((nq >> 3) & 7) - ((nk >> 3) & 7);
            const int dw = (nq & 7) - (nk & 7);
            const int ridx = (dd + 3) * 225 + (dh + 7) * 15 + (dw + 7);
            float val = s.f[v] + rpb[(size_t)ridx * NH + head];
            if (masked && Rg[nq] != Rg[nk]) val -= 100.0f;
            Sc[m][nk] = val;
        }
    }
    __syncthreads();

    // row softmax (16 rows on lanes 0..15), float4-vectorized
    if (lane < 16) {
        const v4f* sr = (const v4f*)&Sc[lane][0];
        float mx = -3.0e38f;
        for (int j = 0; j < 64; ++j) {
            const v4f s = sr[j];
            mx = fmaxf(mx, fmaxf(fmaxf(s[0], s[1]), fmaxf(s[2], s[3])));
        }
        float sum = 0.0f;
        for (int j = 0; j < 64; ++j) {
            const v4f s = sr[j];
            const float e0 = __expf(s[0] - mx), e1 = __expf(s[1] - mx);
            const float e2 = __expf(s[2] - mx), e3 = __expf(s[3] - mx);
            sum += (e0 + e1) + (e2 + e3);
            Sc[lane][4 * j + 0] = e0; Sc[lane][4 * j + 1] = e1;
            Sc[lane][4 * j + 2] = e2; Sc[lane][4 * j + 3] = e3;
        }
        const float inv = 1.0f / sum;
        for (int j = 0; j < 64; ++j) {
            const v4f s = *(const v4f*)&Sc[lane][4 * j];
            _Float16* pd = &Ps[lane][4 * j];
            pd[0] = (_Float16)(s[0] * inv); pd[1] = (_Float16)(s[1] * inv);
            pd[2] = (_Float16)(s[2] * inv); pd[3] = (_Float16)(s[3] * inv);
        }
    }
    __syncthreads();

    // out = P(16x256) @ V(256x32): 2 col tiles x 8 K-chunks of 32
    for (int ct2 = 0; ct2 < 2; ++ct2) {
        CFrag o;
        #pragma unroll
        for (int i = 0; i < 8; ++i) o.f[i] = 0.0f;
        #pragma unroll
        for (int kt = 0; kt < 8; ++kt) {
            AFrag a, b;
            const _Float16* pr = &Ps[row][kt * 32];
            a.g[0] = *(const v8h*)(pr + half * 8);
            a.g[1] = *(const v8h*)(pr + 16 + half * 8);
            const _Float16* vr = &VsT[ct2 * 16 + row][kt * 32];
            b.g[0] = *(const v8h*)(vr + half * 8);
            b.g[1] = *(const v8h*)(vr + 16 + half * 8);
            o.v = __builtin_amdgcn_wmma_f32_16x16x32_f16(
                false, a.v, false, b.v, (short)0, o.v, false, false);
        }
        #pragma unroll
        for (int v = 0; v < 8; ++v) {
            const int m  = v + 8 * half;
            const int nq = rt * 16 + m;
            const int t = tok_index(nq, wdi, whi, wwi, Dd, Hh, Ww, sD, sH, sW);
            out[(size_t)t * C + head * 32 + ct2 * 16 + row] = o.f[v];
        }
    }
}

// ---------------------------------------------------------------------------
// LayerNorm over last dim C; one block (128 thr) per row. eps = 1e-5.
// ---------------------------------------------------------------------------
__global__ __launch_bounds__(128) void layernorm(
    const float* __restrict__ x, const float* __restrict__ g,
    const float* __restrict__ b, float* __restrict__ y, int C)
{
    __shared__ float s1[128], s2[128];
    const float* xr = x + (size_t)blockIdx.x * C;
    float* yr = y + (size_t)blockIdx.x * C;
    float a = 0.0f, q = 0.0f;
    for (int c = threadIdx.x; c < C; c += 128) {
        const float v = xr[c];
        a += v; q += v * v;
    }
    s1[threadIdx.x] = a; s2[threadIdx.x] = q;
    __syncthreads();
    for (int off = 64; off > 0; off >>= 1) {
        if (threadIdx.x < off) {
            s1[threadIdx.x] += s1[threadIdx.x + off];
            s2[threadIdx.x] += s2[threadIdx.x + off];
        }
        __syncthreads();
    }
    const float mean = s1[0] / (float)C;
    const float var  = s2[0] / (float)C - mean * mean;
    const float inv  = rsqrtf(var + 1e-5f);
    for (int c = threadIdx.x; c < C; c += 128)
        yr[c] = (xr[c] - mean) * inv * g[c] + b[c];
}

// ---------------------------------------------------------------------------
// Patch embedding im2col: (1,5,32,256,256) -> G[32768, 320]
// feature f = c*64 + pd*16 + ph*4 + pw
// ---------------------------------------------------------------------------
__global__ void im2col_patch(const float* __restrict__ x, float* __restrict__ G)
{
    const size_t i = (size_t)blockIdx.x * 256 + threadIdx.x;
    if (i >= 32768ull * 320ull) return;
    const int f = (int)(i % 320);
    const int tok = (int)(i / 320);
    const int w = tok & 63, h = (tok >> 6) & 63, d = tok >> 12;
    const int c = f >> 6, r = f & 63;
    const int pd = r >> 4, ph = (r >> 2) & 3, pw = r & 3;
    G[i] = x[(((size_t)c * 32 + (d * 4 + pd)) * 256 + (h * 4 + ph)) * 256
             + (w * 4 + pw)];
}

// ---------------------------------------------------------------------------
// Patch merging gather: x[D,H,W,C] -> y[D,H/2,W/2, 4C]  (concat x0|x1|x2|x3)
// part p -> (dh = p&1, dw = p>>1)
// ---------------------------------------------------------------------------
__global__ void merge_gather(const float* __restrict__ x, float* __restrict__ y,
                             int Dd, int Hh, int Ww, int C)
{
    const int H2 = Hh >> 1, W2 = Ww >> 1, C4 = 4 * C;
    const size_t total = (size_t)Dd * H2 * W2 * C4;
    const size_t i = (size_t)blockIdx.x * 256 + threadIdx.x;
    if (i >= total) return;
    const int f = (int)(i % C4);
    const size_t tok = i / C4;
    const int w = (int)(tok % W2);
    const int h = (int)((tok / W2) % H2);
    const int d = (int)(tok / ((size_t)W2 * H2));
    const int part = f / C, c = f % C;
    const int dh = part & 1, dw = part >> 1;
    y[i] = x[(((size_t)d * Hh + (2 * h + dh)) * Ww + (2 * w + dw)) * C + c];
}

// ---------------------------------------------------------------------------
// Mean over tokens: out[c] = mean_t y[t,c]
// ---------------------------------------------------------------------------
__global__ void mean_tokens(const float* __restrict__ y, float* __restrict__ out,
                            int T, int C)
{
    const int c = blockIdx.x * blockDim.x + threadIdx.x;
    if (c >= C) return;
    float s = 0.0f;
    for (int t = 0; t < T; ++t) s += y[(size_t)t * C + c];
    out[c] = s / (float)T;
}

// ---------------------------------------------------------------------------
// Host orchestration. Input order = JAX pytree flatten (sorted dict keys):
//  [0..8]  merges: (nb, ng, red_w) x3
//  [9]     norm_b   [10] norm_g
//  [11]    pe_b  [12] pe_nb  [13] pe_ng  [14] pe_w
//  [15...] 12 blocks x 13 tensors, alphabetical:
//    fc1_b, fc1_w, fc2_b, fc2_w, n1b, n1g, n2b, n2g,
//    proj_b, proj_w, qkv_b, qkv_w, rpb_table
//  [last]  x
// ---------------------------------------------------------------------------
extern "C" void kernel_launch(void* const* d_in, const int* in_sizes, int n_in,
                              void* d_out, int out_size, void* d_ws, size_t ws_size,
                              hipStream_t stream)
{
    (void)in_sizes; (void)out_size; (void)ws_size;
    const float* const* in = (const float* const*)d_in;

    const float* merge_nb[3], *merge_ng[3], *merge_rw[3];
    for (int i = 0; i < 3; ++i) {
        merge_nb[i] = in[i * 3 + 0];
        merge_ng[i] = in[i * 3 + 1];
        merge_rw[i] = in[i * 3 + 2];
    }
    const float* norm_b = in[9];
    const float* norm_g = in[10];
    const float* pe_b  = in[11];
    const float* pe_nb = in[12];
    const float* pe_ng = in[13];
    const float* pe_w  = in[14];
    const float* x_in  = in[n_in - 1];

    const int depths[4] = {2, 2, 6, 2};
    const int heads [4] = {3, 6, 12, 24};
    int bbase[4]; { int a = 0; for (int s = 0; s < 4; ++s) { bbase[s] = 15 + a * 13; a += depths[s]; } }

    // workspace partition (floats)
    float* ws = (float*)d_ws;
    float* bX = ws;                    // T*C        (max 32768*96)
    float* bY = bX + 3145728;          // LN out / merge gather
    float* bQ = bY + 3145728;          // T*3C       (max 32768*288)
    float* bA = bQ + 9437184;          // attention out T*C
    float* bH = bA + 3145728;          // MLP hidden T*4C / im2col (max 32768*384)

    // ---- patch embed: im2col -> GEMM(320->96) -> LN ----
    {
        const size_t total = 32768ull * 320ull;
        im2col_patch<<<(unsigned)((total + 255) / 256), 256, 0, stream>>>(x_in, bH);
        gemm_wmma<<<dim3(96 / 32, 32768 / 64), 128, 0, stream>>>(
            bH, pe_w, pe_b, nullptr, bX, 32768, 96, 320, 0);
        layernorm<<<32768, 128, 0, stream>>>(bX, pe_ng, pe_nb, bX, 96);
    }

    int D = 8, H = 64, W = 64, C = 96;
    for (int si = 0; si < 4; ++si) {
        const int NH = heads[si];
        const int T = D * H * W;
        for (int bi = 0; bi < depths[si]; ++bi) {
            const float* const* bp = in + bbase[si] + bi * 13;
            const float* fc1_b = bp[0],  *fc1_w = bp[1];
            const float* fc2_b = bp[2],  *fc2_w = bp[3];
            const float* n1b = bp[4],    *n1g = bp[5];
            const float* n2b = bp[6],    *n2g = bp[7];
            const float* proj_b = bp[8], *proj_w = bp[9];
            const float* qkv_b = bp[10], *qkv_w = bp[11];
            const float* rpb = bp[12];

            const int sD = (bi & 1) ? 2 : 0;                  // D=8 > win 4 always
            const int sH = ((bi & 1) && H > 8) ? 4 : 0;
            const int sW = ((bi & 1) && W > 8) ? 4 : 0;

            // LN1 -> QKV
            layernorm<<<T, 128, 0, stream>>>(bX, n1g, n1b, bY, C);
            gemm_wmma<<<dim3((3 * C) / 32, T / 64), 128, 0, stream>>>(
                bY, qkv_w, qkv_b, nullptr, bQ, T, 3 * C, C, 0);

            // windowed attention
            const int nW = (D / 4) * (H / 8) * (W / 8);
            attn_win<<<nW * NH * 16, 32, 0, stream>>>(
                bQ, rpb, bA, D, H, W, C, NH, sD, sH, sW);

            // proj + residual
            gemm_wmma<<<dim3(C / 32, T / 64), 128, 0, stream>>>(
                bA, proj_w, proj_b, bX, bX, T, C, C, 0);

            // LN2 -> MLP (fc1 + GELU, fc2 + residual)
            layernorm<<<T, 128, 0, stream>>>(bX, n2g, n2b, bY, C);
            gemm_wmma<<<dim3((4 * C) / 32, T / 64), 128, 0, stream>>>(
                bY, fc1_w, fc1_b, nullptr, bH, T, 4 * C, C, 1);
            gemm_wmma<<<dim3(C / 32, T / 64), 128, 0, stream>>>(
                bH, fc2_w, fc2_b, bX, bX, T, C, 4 * C, 0);
        }
        if (si < 3) { // patch merge (H,W halved, C doubled)
            const int H2 = H / 2, W2 = W / 2, T2 = D * H2 * W2, C4 = 4 * C;
            const size_t total = (size_t)T2 * C4;
            merge_gather<<<(unsigned)((total + 255) / 256), 256, 0, stream>>>(
                bX, bY, D, H, W, C);
            layernorm<<<T2, 128, 0, stream>>>(bY, merge_ng[si], merge_nb[si], bY, C4);
            gemm_wmma<<<dim3((2 * C) / 32, T2 / 64), 128, 0, stream>>>(
                bY, merge_rw[si], nullptr, nullptr, bX, T2, 2 * C, C4, 0);
            H = H2; W = W2; C *= 2;
        }
    }

    // final LN + global mean -> out[768]
    const int Tf = D * H * W; // 512
    layernorm<<<Tf, 128, 0, stream>>>(bX, norm_g, norm_b, bY, C);
    mean_tokens<<<(C + 255) / 256, 256, 0, stream>>>(bY, (float*)d_out, Tf, C);
}
